// ALayer_v2_14637248545083
// MI455X (gfx1250) — matrix operand order repaired and checked by reading, not compiled, not run on hardware
//
#include <hip/hip_runtime.h>
#include <hip/hip_bf16.h>

typedef __attribute__((ext_vector_type(16))) __bf16 v16bf;
typedef __attribute__((ext_vector_type(8)))  __bf16 v8bf;
typedef __attribute__((ext_vector_type(4)))  __bf16 v4bf;
typedef __attribute__((ext_vector_type(8)))  float  v8f;

// ---------------------------------------------------------------------------
// Kernel 1: conv1 (256 -> 16 ch, 3x3, pad 1) + ReLU via bf16 WMMA.
// One block per (b, y) row; 128 threads = 4 waves; wave w owns pixels
// [16w, 16w+16).  D = 16 out-ch x 16 px, f32 accumulate.
//
// Hot loop is branch-free: input row staged in LDS (bf16, 8-elem left pad so
// every 16-px window is 16B aligned, zeroed halos), weights staged in the
// WMMA A-fragment layout for all 3 dx of the current dy.  Per (dy, cc):
// 3x16 scalar bf16 LDS reads (16 ch, dx -1/0/+1) + 3 A-fragment loads + 3 WMMAs.
// 72 x v_wmma_f32_16x16x32_bf16 per wave total.
// ---------------------------------------------------------------------------
#define XSTRIDE 72   // bf16 elems per channel row in LDS (144 B, 16B-multiple)

__global__ void __launch_bounds__(128)
aconv1_wmma_kernel(const float* __restrict__ x_in,
                   const float* __restrict__ w1,
                   float* __restrict__ tmp1)
{
    __shared__ __align__(16) __bf16 ldsX[256 * XSTRIDE + 8]; // 36.9 KB
    __shared__ __align__(32) __bf16 ldsW[3 * 8 * 32 * 16];   // 24 KB

    const int tid  = threadIdx.x;
    const int lane = tid & 31;
    const int wave = tid >> 5;
    const int y    = blockIdx.x & 63;
    const int b    = blockIdx.x >> 6;
    const int x0   = wave << 4;

    // Zero the halo pads once; the per-dy fills only touch the data region
    // (elems 8..71 of each channel row), so pads persist.
    for (int j = tid; j < 256 * 8 + 8; j += 128) {
        if (j < 2048) ldsX[(j >> 3) * XSTRIDE + (j & 7)] = (__bf16)0.0f;
        else          ldsX[256 * XSTRIDE + (j - 2048)]   = (__bf16)0.0f;
    }

    v8f acc = {0.f, 0.f, 0.f, 0.f, 0.f, 0.f, 0.f, 0.f};

    for (int dyI = 0; dyI < 3; ++dyI) {
        const int yy = y + dyI - 1;
        if ((unsigned)yy >= 64u) continue;          // block-uniform skip

        __syncthreads();  // previous-stage readers done before overwrite

        // ---- stage input row: 256 ch x 64 px -> bf16, coalesced float4 ----
        for (int j = tid; j < 4096; j += 128) {
            const int c  = j >> 4;
            const int xq = (j & 15) << 2;
            const float* src =
                x_in + (size_t)((((b << 8) + c) << 6) + yy) * 64 + xq;
            __builtin_prefetch(src + 64, 0, 0);      // next row, same channel
            const float4 f = *(const float4*)src;
            v4bf v;
            v[0] = (__bf16)f.x; v[1] = (__bf16)f.y;
            v[2] = (__bf16)f.z; v[3] = (__bf16)f.w;
            *(v4bf*)&ldsX[c * XSTRIDE + 8 + xq] = v;
        }

        // ---- stage weights for the 3 dx of this dy in A-fragment layout ---
        // 16-bit A 16x32 (per ISA): lane<16 K=8g+2v+h (v<4) / 16+8g+2(v-4)+h
        for (int j = tid; j < 3 * 4096; j += 128) {
            const int dxI = j >> 12;
            const int r   = j & 4095;
            const int cc  = r >> 9;
            const int ln  = (r >> 4) & 31;
            const int t   = r & 15;              // t = 2v + h
            const int m   = ln & 15;
            const int g   = ln >> 4;
            const int v   = t >> 1;
            const int h   = t & 1;
            const int k   = (v < 4) ? (g * 8 + 2 * v + h)
                                    : (16 + g * 8 + 2 * (v - 4) + h);
            ldsW[j] = (__bf16)w1[(m * 256 + cc * 32 + k) * 9 + dyI * 3 + dxI];
        }
        __syncthreads();

        // ---- compute: branch-free, EXEC all-ones ----
#pragma unroll
        for (int cc = 0; cc < 8; ++cc) {
            const int bn = lane & 15;
            const int bh = lane >> 4;

            v16bf w0, bm, bp;                          // dx = 0 / -1 / +1
#pragma unroll
            for (int s = 0; s < 16; ++s) {
                const int kk = (s < 8) ? (8 * bh + s) : (16 + 8 * bh + (s - 8));
                const int p  = (cc * 32 + kk) * XSTRIDE + 8 + x0 + bn;
                w0[s] = ldsX[p];
                bm[s] = ldsX[p - 1];
                bp[s] = ldsX[p + 1];
            }

            const v16bf a0 = *(const v16bf*)&ldsW[((0 * 8 + cc) * 32 + lane) << 4];
            const v16bf a1 = *(const v16bf*)&ldsW[((1 * 8 + cc) * 32 + lane) << 4];
            const v16bf a2 = *(const v16bf*)&ldsW[((2 * 8 + cc) * 32 + lane) << 4];

            acc = __builtin_amdgcn_wmma_f32_16x16x32_bf16(
                false, a0, false, bm, (short)0, acc, false, false);
            acc = __builtin_amdgcn_wmma_f32_16x16x32_bf16(
                false, a1, false, w0, (short)0, acc, false, false);
            acc = __builtin_amdgcn_wmma_f32_16x16x32_bf16(
                false, a2, false, bp, (short)0, acc, false, false);
        }
    }

    // D layout: lane -> N = lane&15, M = r + 8*(lane>>4).  ReLU + store.
    const int n   = lane & 15;
    const int mhi = (lane >> 4) << 3;
#pragma unroll
    for (int r = 0; r < 8; ++r) {
        const int m = r + mhi;
        float v = acc[r];
        v = v > 0.f ? v : 0.f;
        tmp1[((((b << 4) + m) << 6) + y) * 64 + x0 + n] = v;
    }
}

// ---------------------------------------------------------------------------
// Kernel 2: conv2 (16 -> 1 ch, 3x3, pad 1) + sigmoid.  One thread per pixel.
// ---------------------------------------------------------------------------
__global__ void aconv2_sigmoid_kernel(const float* __restrict__ tmp1,
                                      const float* __restrict__ w2,
                                      float* __restrict__ A)
{
    const int p = blockIdx.x * blockDim.x + threadIdx.x;
    if (p >= 32 * 64 * 64) return;
    const int x = p & 63;
    const int y = (p >> 6) & 63;
    const int b = p >> 12;

    float acc = 0.f;
    for (int c = 0; c < 16; ++c) {
        const float* base = tmp1 + (((b << 4) + c) << 12);
#pragma unroll
        for (int ky = 0; ky < 3; ++ky) {
            const int yy = y + ky - 1;
            if ((unsigned)yy >= 64u) continue;
#pragma unroll
            for (int kx = 0; kx < 3; ++kx) {
                const int xx = x + kx - 1;
                if ((unsigned)xx >= 64u) continue;
                acc += base[(yy << 6) + xx] * w2[c * 9 + ky * 3 + kx];
            }
        }
    }
    A[p] = 1.0f / (1.0f + __expf(-acc));
}

// ---------------------------------------------------------------------------
// Kernel 3: 3x3 zero-padded box-sum of A.
// ---------------------------------------------------------------------------
__global__ void abox3_kernel(const float* __restrict__ A,
                             float* __restrict__ A_box)
{
    const int p = blockIdx.x * blockDim.x + threadIdx.x;
    if (p >= 32 * 64 * 64) return;
    const int x = p & 63;
    const int y = (p >> 6) & 63;
    const int b = p >> 12;

    const float* base = A + (b << 12);
    float s = 0.f;
#pragma unroll
    for (int ky = 0; ky < 3; ++ky) {
        const int yy = y + ky - 1;
        if ((unsigned)yy >= 64u) continue;
#pragma unroll
        for (int kx = 0; kx < 3; ++kx) {
            const int xx = x + kx - 1;
            if ((unsigned)xx >= 64u) continue;
            s += base[(yy << 6) + xx];
        }
    }
    A_box[p] = s;
}

// ---------------------------------------------------------------------------
// Kernel 4: out = x_out * A_box (broadcast over 256 channels).
// float4 streaming pass; A_box (512 KiB) stays L2-resident.
// ---------------------------------------------------------------------------
__global__ void ascale_kernel(const float* __restrict__ x_out,
                              const float* __restrict__ A_box,
                              float* __restrict__ out)
{
    const long long t = (long long)blockIdx.x * blockDim.x + threadIdx.x;
    const long long e = t << 2;
    const int x = (int)(e & 63);
    const int y = (int)((e >> 6) & 63);
    const int b = (int)(e >> 20);

    const float4 xo = *(const float4*)(x_out + e);
    const float4 ab = *(const float4*)(A_box + ((((b << 6) + y) << 6) + x));
    float4 r;
    r.x = xo.x * ab.x;
    r.y = xo.y * ab.y;
    r.z = xo.z * ab.z;
    r.w = xo.w * ab.w;
    *(float4*)(out + e) = r;
}

// ---------------------------------------------------------------------------
extern "C" void kernel_launch(void* const* d_in, const int* in_sizes, int n_in,
                              void* d_out, int out_size, void* d_ws, size_t ws_size,
                              hipStream_t stream)
{
    const float* x_in  = (const float*)d_in[0];   // [32,256,64,64]
    const float* x_out = (const float*)d_in[1];   // [32,256,64,64]
    const float* w1    = (const float*)d_in[2];   // [16,256,3,3]
    const float* w2    = (const float*)d_in[3];   // [1,16,3,3]
    // d_in[4], d_in[5] (fc1, fc2) are dead code in the reference.
    float* out = (float*)d_out;

    char*  ws   = (char*)d_ws;
    float* tmp1 = (float*)ws;                                   // 8 MiB
    float* A    = (float*)(ws + (size_t)32 * 16 * 64 * 64 * 4); // 512 KiB
    float* Abox = A + 32 * 64 * 64;                             // 512 KiB

    aconv1_wmma_kernel<<<32 * 64, 128, 0, stream>>>(x_in, w1, tmp1);
    aconv2_sigmoid_kernel<<<512, 256, 0, stream>>>(tmp1, w2, A);
    abox3_kernel<<<512, 256, 0, stream>>>(A, Abox);
    ascale_kernel<<<32768, 256, 0, stream>>>(x_out, Abox, out);
}
